// TransformerLayer_3040836846037
// MI455X (gfx1250) — compile-verified, matrix operand order checked by reading, not run on hardware
//
#include <hip/hip_runtime.h>

// ---------------------------------------------------------------------------
// TransformerLayer for MI455X (gfx1250): bf16 WMMA GEMMs + flash attention.
// B=2, S=2048, D=768, H=12, Dk=64, Dff=3072.  All matmuls use
// v_wmma_f32_16x16x32_bf16; activations/weights converted to bf16, f32 accum.
// Tile staging uses GLOBAL_LOAD_ASYNC_TO_LDS_B128 (ASYNCcnt) when available,
// double-buffered so copies overlap WMMA.
// ---------------------------------------------------------------------------

typedef __attribute__((ext_vector_type(16))) __bf16        v16bf;
typedef __attribute__((ext_vector_type(8)))  float         v8f;
typedef __attribute__((ext_vector_type(4)))  unsigned int  u32x4;

union ABfrag { v16bf v; u32x4 q[2]; };

#if defined(__HIP_DEVICE_COMPILE__) && \
    __has_builtin(__builtin_amdgcn_global_load_async_to_lds_b128) && \
    __has_builtin(__builtin_amdgcn_s_wait_asynccnt)
#define USE_ASYNC 1
typedef __attribute__((__vector_size__(4 * sizeof(int)))) int i32x4;
typedef __attribute__((address_space(1))) i32x4* gas_b128_t;   // global src
typedef __attribute__((address_space(3))) i32x4* las_b128_t;   // LDS dst

__device__ __forceinline__ void async_cp_b128(const void* gsrc, void* ldst) {
    void* g = const_cast<void*>(gsrc);
    __builtin_amdgcn_global_load_async_to_lds_b128(
        (gas_b128_t)g, (las_b128_t)ldst, 0, 0);
}
#else
#define USE_ASYNC 0
#endif

__device__ __forceinline__ v8f v8f_zero() {
    v8f z;
#pragma unroll
    for (int i = 0; i < 8; ++i) z[i] = 0.0f;
    return z;
}

// float -> bf16 (round-to-nearest-even), stored as u16 to avoid scalar __bf16 ops
__device__ __forceinline__ unsigned short f2bf(float f) {
    unsigned u = __builtin_bit_cast(unsigned, f);
    if ((u & 0x7FFFFFFFu) > 0x7F800000u)          // NaN
        return (unsigned short)((u >> 16) | 0x40);
    unsigned r = u + 0x7FFFu + ((u >> 16) & 1u);
    return (unsigned short)(r >> 16);
}

#define D_MODEL 768
#define N_HEAD  12
#define D_K     64
#define D_FF    3072
#define S_LEN   2048
#define M_ROWS  4096   // B*S

// ---------------------------------------------------------------------------
__global__ void cvt_bf16_kernel(const float* __restrict__ in,
                                unsigned short* __restrict__ out, int n) {
    int i = blockIdx.x * 256 + threadIdx.x;
    if (i < n) out[i] = f2bf(in[i]);
}

// W[K][N] f32 -> WT[N][K] bf16
__global__ void transpose_cvt_kernel(const float* __restrict__ W,
                                     unsigned short* __restrict__ WT,
                                     int K, int N) {
    int n = blockIdx.x * 16 + threadIdx.x;
    int k = blockIdx.y * 16 + threadIdx.y;
    if (k < K && n < N) WT[(size_t)n * K + k] = f2bf(W[(size_t)k * N + n]);
}

// ---------------------------------------------------------------------------
// GEMM: out[M][N] = A[M][K](bf16) * WT[N][K]^T(bf16) + bias, epilogues:
//   0: store bf16    1: relu + store bf16    2: +resid, store f32
// Block 256 thr (8 waves), tile 128x128, K-step 32, async double-buffered LDS.
// ---------------------------------------------------------------------------
template <int EPI>
__global__ __launch_bounds__(256)
void gemm_bf16_wmma(const unsigned short* __restrict__ A,
                    const unsigned short* __restrict__ WT,
                    const float* __restrict__ bias,
                    const float* __restrict__ resid,
                    float* __restrict__ outF,
                    unsigned short* __restrict__ outB,
                    int M, int N, int K) {
#if USE_ASYNC
    __shared__ unsigned short sA[2][128 * 32];
    __shared__ unsigned short sB[2][128 * 32];
#else
    __shared__ unsigned short sA[1][128 * 32];
    __shared__ unsigned short sB[1][128 * 32];
#endif
    const int tid  = threadIdx.x;
    const int lane = tid & 31, l16 = lane & 15, half = lane >> 4;
    const int wid  = tid >> 5;
    const int wm   = wid & 1;        // 2 waves along M (64 rows each)
    const int wn   = wid >> 1;       // 4 waves along N (32 cols each)
    const int m0   = blockIdx.y * 128, n0 = blockIdx.x * 128;

    v8f acc[4][2];
#pragma unroll
    for (int i = 0; i < 4; ++i)
#pragma unroll
        for (int j = 0; j < 2; ++j) acc[i][j] = v8f_zero();

    // preload all fragments, then 8 back-to-back WMMAs (distinct accumulators)
    auto compute_tile = [&](const unsigned short* tA, const unsigned short* tB) {
        ABfrag af[4], bf[2];
#pragma unroll
        for (int i = 0; i < 4; ++i) {
            const unsigned short* p = &tA[(wm * 64 + i * 16 + l16) * 32];
            af[i].q[0] = *(const u32x4*)(p + half * 8);
            af[i].q[1] = *(const u32x4*)(p + 16 + half * 8);
        }
#pragma unroll
        for (int j = 0; j < 2; ++j) {
            const unsigned short* p = &tB[(wn * 32 + j * 16 + l16) * 32];
            bf[j].q[0] = *(const u32x4*)(p + half * 8);
            bf[j].q[1] = *(const u32x4*)(p + 16 + half * 8);
        }
#pragma unroll
        for (int i = 0; i < 4; ++i)
#pragma unroll
            for (int j = 0; j < 2; ++j)
                acc[i][j] = __builtin_amdgcn_wmma_f32_16x16x32_bf16(
                    false, af[i].v, false, bf[j].v, (short)0, acc[i][j],
                    false, false);
    };

    const int nt = K >> 5;
#if USE_ASYNC
    // each thread copies 2 A segs + 2 B segs (16B each) per tile, async
    auto issue_tile = [&](int k0, int buf) {
#pragma unroll
        for (int i = 0; i < 2; ++i) {
            int seg = tid + i * 256;
            int row = seg >> 2, kk = (seg & 3) << 3;
            async_cp_b128(&A[(size_t)(m0 + row) * K + k0 + kk],
                          &sA[buf][row * 32 + kk]);
            async_cp_b128(&WT[(size_t)(n0 + row) * K + k0 + kk],
                          &sB[buf][row * 32 + kk]);
        }
    };
    issue_tile(0, 0);
    for (int kt = 0; kt < nt; ++kt) {
        const int buf = kt & 1;
        if (kt + 1 < nt) {
            issue_tile((kt + 1) << 5, buf ^ 1);     // prefetch next tile
            __builtin_amdgcn_s_wait_asynccnt(4);    // current tile landed
        } else {
            __builtin_amdgcn_s_wait_asynccnt(0);
        }
        __syncthreads();
        compute_tile(sA[buf], sB[buf]);
        __syncthreads();                            // done reading before reuse
    }
#else
    for (int kt = 0; kt < nt; ++kt) {
        const int k0 = kt << 5;
        __syncthreads();
#pragma unroll
        for (int i = 0; i < 2; ++i) {
            int seg = tid + i * 256;
            int row = seg >> 2, kk = (seg & 3) << 3;
            *(u32x4*)&sA[0][row * 32 + kk] =
                *(const u32x4*)&A[(size_t)(m0 + row) * K + k0 + kk];
            *(u32x4*)&sB[0][row * 32 + kk] =
                *(const u32x4*)&WT[(size_t)(n0 + row) * K + k0 + kk];
        }
        __syncthreads();
        compute_tile(sA[0], sB[0]);
    }
#endif

    // epilogue: C layout -> lane holds col n=l16, rows m = r + 8*half
#pragma unroll
    for (int j = 0; j < 2; ++j) {
        int n = n0 + wn * 32 + j * 16 + l16;
        float bv = bias[n];
#pragma unroll
        for (int i = 0; i < 4; ++i) {
#pragma unroll
            for (int r = 0; r < 8; ++r) {
                int m = m0 + wm * 64 + i * 16 + half * 8 + r;
                size_t idx = (size_t)m * N + n;
                float v = acc[i][j][r] + bv;
                if (EPI == 1) { v = v > 0.f ? v : 0.f; outB[idx] = f2bf(v); }
                else if (EPI == 0) { outB[idx] = f2bf(v); }
                else { outF[idx] = v + resid[idx]; }
            }
        }
    }
}

// ---------------------------------------------------------------------------
// Flash attention: one block = 64 query rows of one (b,h).  Block 128 thr
// (4 waves, 16 rows each).  KV streamed in 64-key tiles.  K tile copied with
// async-to-LDS (overlaps the manual V transpose staging).
// ---------------------------------------------------------------------------
__global__ __launch_bounds__(128)
void flash_attn_kernel(const unsigned short* __restrict__ Qm,
                       const unsigned short* __restrict__ Km,
                       const unsigned short* __restrict__ Vm,
                       unsigned short* __restrict__ Om) {
    __shared__ unsigned short sK[64 * 64];       // [kv][d]
    __shared__ unsigned short sVT[64 * 80];      // [d][kv], pad->16B aligned rows
    __shared__ unsigned short sP[4][16 * 64];    // per-wave P tile

    const int tid  = threadIdx.x;
    const int lane = tid & 31, l16 = lane & 15, half = lane >> 4;
    const int w    = tid >> 5;
    const int bh   = blockIdx.y;
    const int b    = bh / N_HEAD, h = bh % N_HEAD;
    const int q0   = blockIdx.x * 64;
    const size_t rowbase = (size_t)b * S_LEN;
    const int hoff = h * D_K;
    const float scale = 0.125f;                  // 1/sqrt(64)

    // Q fragments pinned in registers (16 rows x 64 d per wave)
    ABfrag qf[2];
    {
        const unsigned short* qp =
            Qm + (rowbase + q0 + w * 16 + l16) * D_MODEL + hoff;
        qf[0].q[0] = *(const u32x4*)(qp + half * 8);
        qf[0].q[1] = *(const u32x4*)(qp + 16 + half * 8);
        qf[1].q[0] = *(const u32x4*)(qp + 32 + half * 8);
        qf[1].q[1] = *(const u32x4*)(qp + 48 + half * 8);
    }

    v8f ctx[4];
#pragma unroll
    for (int jj = 0; jj < 4; ++jj) ctx[jj] = v8f_zero();
    float mrow[8], lrow[8];
#pragma unroll
    for (int r = 0; r < 8; ++r) { mrow[r] = -1e30f; lrow[r] = 0.f; }

    for (int kv0 = 0; kv0 < S_LEN; kv0 += 64) {
        __syncthreads();
        // ---- K tile: 512 segs of 8 bf16 -> sK[kv][d]
#pragma unroll
        for (int i = 0; i < 4; ++i) {
            int seg = tid + i * 128;
            int row = seg >> 3, kk = (seg & 7) << 3;
#if USE_ASYNC
            async_cp_b128(&Km[(rowbase + kv0 + row) * D_MODEL + hoff + kk],
                          &sK[row * 64 + kk]);
#else
            *(u32x4*)&sK[row * 64 + kk] =
                *(const u32x4*)&Km[(rowbase + kv0 + row) * D_MODEL + hoff + kk];
#endif
        }
        // ---- V tile, transposed into sVT[d][kv] (overlaps async K copy)
#pragma unroll
        for (int i = 0; i < 4; ++i) {
            int seg = tid + i * 128;
            int row = seg >> 3, kk = (seg & 7) << 3;
            union { u32x4 q; unsigned short e[8]; } t;
            t.q = *(const u32x4*)&Vm[(rowbase + kv0 + row) * D_MODEL + hoff + kk];
#pragma unroll
            for (int e = 0; e < 8; ++e) sVT[(kk + e) * 80 + row] = t.e[e];
        }
#if USE_ASYNC
        __builtin_amdgcn_s_wait_asynccnt(0);
#endif
        __syncthreads();

        // ---- scores S = Q K^T : preload all 8 K fragments, then 8 WMMAs
        v8f sc[4];
#pragma unroll
        for (int j = 0; j < 4; ++j) sc[j] = v8f_zero();
        {
            ABfrag kf[4][2];
#pragma unroll
            for (int j = 0; j < 4; ++j)
#pragma unroll
                for (int c = 0; c < 2; ++c) {
                    const unsigned short* p = &sK[(j * 16 + l16) * 64 + c * 32];
                    kf[j][c].q[0] = *(const u32x4*)(p + half * 8);
                    kf[j][c].q[1] = *(const u32x4*)(p + 16 + half * 8);
                }
#pragma unroll
            for (int c = 0; c < 2; ++c)          // c outer: accumulators interleave
#pragma unroll
                for (int j = 0; j < 4; ++j)
                    sc[j] = __builtin_amdgcn_wmma_f32_16x16x32_bf16(
                        false, qf[c].v, false, kf[j][c].v, (short)0, sc[j],
                        false, false);
        }

        // ---- online softmax; row m = r + 8*half lives in vgpr r across the
        // 16 lanes of this half -> xor-reduce with masks < 16.
#pragma unroll
        for (int r = 0; r < 8; ++r) {
            float s0 = sc[0][r] * scale, s1 = sc[1][r] * scale;
            float s2 = sc[2][r] * scale, s3 = sc[3][r] * scale;
            float mx = fmaxf(fmaxf(s0, s1), fmaxf(s2, s3));
#pragma unroll
            for (int d = 1; d < 16; d <<= 1)
                mx = fmaxf(mx, __shfl_xor(mx, d, 32));
            float mnew  = fmaxf(mrow[r], mx);
            float alpha = __expf(mrow[r] - mnew);
            float p0 = __expf(s0 - mnew), p1 = __expf(s1 - mnew);
            float p2 = __expf(s2 - mnew), p3 = __expf(s3 - mnew);
            unsigned short* pr = &sP[w][(half * 8 + r) * 64];
            pr[ 0 + l16] = f2bf(p0);
            pr[16 + l16] = f2bf(p1);
            pr[32 + l16] = f2bf(p2);
            pr[48 + l16] = f2bf(p3);
            float rs = p0 + p1 + p2 + p3;
#pragma unroll
            for (int d = 1; d < 16; d <<= 1) rs += __shfl_xor(rs, d, 32);
            lrow[r] = lrow[r] * alpha + rs;
            mrow[r] = mnew;
#pragma unroll
            for (int jj = 0; jj < 4; ++jj) ctx[jj][r] *= alpha;
        }

        // ---- ctx += P @ V : preload P and all V^T fragments, then 8 WMMAs
        {
            ABfrag pf[2], vf[4][2];
#pragma unroll
            for (int c = 0; c < 2; ++c) {
                const unsigned short* pp = &sP[w][l16 * 64 + c * 32];
                pf[c].q[0] = *(const u32x4*)(pp + half * 8);
                pf[c].q[1] = *(const u32x4*)(pp + 16 + half * 8);
            }
#pragma unroll
            for (int jj = 0; jj < 4; ++jj)
#pragma unroll
                for (int c = 0; c < 2; ++c) {
                    const unsigned short* vp = &sVT[(jj * 16 + l16) * 80 + c * 32];
                    vf[jj][c].q[0] = *(const u32x4*)(vp + half * 8);
                    vf[jj][c].q[1] = *(const u32x4*)(vp + 16 + half * 8);
                }
#pragma unroll
            for (int c = 0; c < 2; ++c)
#pragma unroll
                for (int jj = 0; jj < 4; ++jj)
                    ctx[jj] = __builtin_amdgcn_wmma_f32_16x16x32_bf16(
                        false, pf[c].v, false, vf[jj][c].v, (short)0, ctx[jj],
                        false, false);
        }
    }

#pragma unroll
    for (int jj = 0; jj < 4; ++jj)
#pragma unroll
        for (int r = 0; r < 8; ++r) {
            int m = q0 + w * 16 + half * 8 + r;
            float v = ctx[jj][r] / lrow[r];
            Om[(rowbase + m) * D_MODEL + hoff + jj * 16 + l16] = f2bf(v);
        }
}

// ---------------------------------------------------------------------------
// LayerNorm over rows of 768; writes f32 and optional bf16 copy.
// ---------------------------------------------------------------------------
__global__ __launch_bounds__(256)
void layernorm_kernel(const float* __restrict__ in, const float* __restrict__ g,
                      const float* __restrict__ be, float* __restrict__ outF,
                      unsigned short* __restrict__ outB) {
    __shared__ float red[256];
    const int row = blockIdx.x;
    const int t = threadIdx.x;
    const float* x = in + (size_t)row * D_MODEL;
    float v0 = x[t], v1 = x[t + 256], v2 = x[t + 512];
    red[t] = v0 + v1 + v2;
    __syncthreads();
    for (int o = 128; o > 0; o >>= 1) { if (t < o) red[t] += red[t + o]; __syncthreads(); }
    float mu = red[0] * (1.0f / D_MODEL);
    __syncthreads();
    float d0 = v0 - mu, d1 = v1 - mu, d2 = v2 - mu;
    red[t] = d0 * d0 + d1 * d1 + d2 * d2;
    __syncthreads();
    for (int o = 128; o > 0; o >>= 1) { if (t < o) red[t] += red[t + o]; __syncthreads(); }
    float rstd = rsqrtf(red[0] * (1.0f / D_MODEL) + 1e-5f);
    float y0 = d0 * rstd * g[t]       + be[t];
    float y1 = d1 * rstd * g[t + 256] + be[t + 256];
    float y2 = d2 * rstd * g[t + 512] + be[t + 512];
    float* o = outF + (size_t)row * D_MODEL;
    o[t] = y0; o[t + 256] = y1; o[t + 512] = y2;
    if (outB) {
        unsigned short* ob = outB + (size_t)row * D_MODEL;
        ob[t] = f2bf(y0); ob[t + 256] = f2bf(y1); ob[t + 512] = f2bf(y2);
    }
}

// ---------------------------------------------------------------------------
extern "C" void kernel_launch(void* const* d_in, const int* in_sizes, int n_in,
                              void* d_out, int out_size, void* d_ws, size_t ws_size,
                              hipStream_t stream) {
    const float* x  = (const float*)d_in[0];
    const float* wq = (const float*)d_in[1];  const float* bq = (const float*)d_in[2];
    const float* wk = (const float*)d_in[3];  const float* bk = (const float*)d_in[4];
    const float* wv = (const float*)d_in[5];  const float* bv = (const float*)d_in[6];
    const float* wo = (const float*)d_in[7];  const float* bo = (const float*)d_in[8];
    const float* w1 = (const float*)d_in[9];  const float* b1 = (const float*)d_in[10];
    const float* w2 = (const float*)d_in[11]; const float* b2 = (const float*)d_in[12];
    const float* g1 = (const float*)d_in[13]; const float* be1 = (const float*)d_in[14];
    const float* g2 = (const float*)d_in[15]; const float* be2 = (const float*)d_in[16];
    float* out = (float*)d_out;

    // workspace carve-up (all sizes multiples of 256 B)
    char* p = (char*)d_ws;
    const size_t SZ_ACT_BF = (size_t)M_ROWS * D_MODEL * 2;
    const size_t SZ_W_BF   = (size_t)D_MODEL * D_MODEL * 2;
    const size_t SZ_WF_BF  = (size_t)D_MODEL * D_FF * 2;
    const size_t SZ_ACT_F  = (size_t)M_ROWS * D_MODEL * 4;
    const size_t SZ_FF_BF  = (size_t)M_ROWS * D_FF * 2;

    unsigned short* x_bf   = (unsigned short*)p; p += SZ_ACT_BF;
    unsigned short* q_bf   = (unsigned short*)p; p += SZ_ACT_BF;
    unsigned short* k_bf   = (unsigned short*)p; p += SZ_ACT_BF;
    unsigned short* v_bf   = (unsigned short*)p; p += SZ_ACT_BF;
    unsigned short* ctx_bf = (unsigned short*)p; p += SZ_ACT_BF;
    unsigned short* h_bf   = (unsigned short*)p; p += SZ_ACT_BF;
    unsigned short* wqT    = (unsigned short*)p; p += SZ_W_BF;
    unsigned short* wkT    = (unsigned short*)p; p += SZ_W_BF;
    unsigned short* wvT    = (unsigned short*)p; p += SZ_W_BF;
    unsigned short* woT    = (unsigned short*)p; p += SZ_W_BF;
    unsigned short* w1T    = (unsigned short*)p; p += SZ_WF_BF;
    unsigned short* w2T    = (unsigned short*)p; p += SZ_WF_BF;
    unsigned short* ff1_bf = (unsigned short*)p; p += SZ_FF_BF;
    float* r1 = (float*)p; p += SZ_ACT_F;
    float* hb = (float*)p; p += SZ_ACT_F;
    float* r2 = (float*)p; p += SZ_ACT_F;

    {
        int n = M_ROWS * D_MODEL;
        cvt_bf16_kernel<<<(n + 255) / 256, 256, 0, stream>>>(x, x_bf, n);
    }
    dim3 tb(16, 16);
    transpose_cvt_kernel<<<dim3(D_MODEL / 16, D_MODEL / 16), tb, 0, stream>>>(wq, wqT, D_MODEL, D_MODEL);
    transpose_cvt_kernel<<<dim3(D_MODEL / 16, D_MODEL / 16), tb, 0, stream>>>(wk, wkT, D_MODEL, D_MODEL);
    transpose_cvt_kernel<<<dim3(D_MODEL / 16, D_MODEL / 16), tb, 0, stream>>>(wv, wvT, D_MODEL, D_MODEL);
    transpose_cvt_kernel<<<dim3(D_MODEL / 16, D_MODEL / 16), tb, 0, stream>>>(wo, woT, D_MODEL, D_MODEL);
    transpose_cvt_kernel<<<dim3(D_FF / 16, D_MODEL / 16), tb, 0, stream>>>(w1, w1T, D_MODEL, D_FF);
    transpose_cvt_kernel<<<dim3(D_MODEL / 16, D_FF / 16), tb, 0, stream>>>(w2, w2T, D_FF, D_MODEL);

    dim3 gQKV(D_MODEL / 128, M_ROWS / 128);
    gemm_bf16_wmma<0><<<gQKV, 256, 0, stream>>>(x_bf, wqT, bq, nullptr, nullptr, q_bf, M_ROWS, D_MODEL, D_MODEL);
    gemm_bf16_wmma<0><<<gQKV, 256, 0, stream>>>(x_bf, wkT, bk, nullptr, nullptr, k_bf, M_ROWS, D_MODEL, D_MODEL);
    gemm_bf16_wmma<0><<<gQKV, 256, 0, stream>>>(x_bf, wvT, bv, nullptr, nullptr, v_bf, M_ROWS, D_MODEL, D_MODEL);

    flash_attn_kernel<<<dim3(S_LEN / 64, 2 * N_HEAD), 128, 0, stream>>>(q_bf, k_bf, v_bf, ctx_bf);

    gemm_bf16_wmma<2><<<gQKV, 256, 0, stream>>>(ctx_bf, woT, bo, x, r1, nullptr, M_ROWS, D_MODEL, D_MODEL);

    layernorm_kernel<<<M_ROWS, 256, 0, stream>>>(r1, g1, be1, hb, h_bf);

    gemm_bf16_wmma<1><<<dim3(D_FF / 128, M_ROWS / 128), 256, 0, stream>>>(h_bf, w1T, b1, nullptr, nullptr, ff1_bf, M_ROWS, D_FF, D_MODEL);

    gemm_bf16_wmma<2><<<gQKV, 256, 0, stream>>>(ff1_bf, w2T, b2, hb, r2, nullptr, M_ROWS, D_MODEL, D_FF);

    layernorm_kernel<<<M_ROWS, 256, 0, stream>>>(r2, g2, be2, out, nullptr);
}
